// GateLoopTransformer_82755429859996
// MI455X (gfx1250) — matrix operand order repaired and checked by reading, not compiled, not run on hardware
//
#include <hip/hip_runtime.h>
#include <hip/hip_bf16.h>

// ---------------------------------------------------------------------------
// Types for CDNA5 WMMA + TDM
// ---------------------------------------------------------------------------
typedef __attribute__((ext_vector_type(16))) __bf16       v16bf;
typedef __attribute__((ext_vector_type(8)))  float        v8f;
typedef __attribute__((ext_vector_type(4)))  unsigned int u32x4;
typedef __attribute__((ext_vector_type(8)))  int          i32x8;
typedef __attribute__((ext_vector_type(4)))  int          i32x4;

#define GEMM_TB 128   // 4 waves of 32
#define BM 64
#define BN 128
#define BK 32
#define LDT 34        // LDS row stride in ushorts (68B: 64B tile row + 4B TDM pad)

__device__ __forceinline__ unsigned short f2bf(float f) {
  unsigned u = __float_as_uint(f);
  u += 0x7FFFu + ((u >> 16) & 1u);      // round-to-nearest-even
  return (unsigned short)(u >> 16);
}

__device__ __forceinline__ float gelu_tanh(float x) {
  float x3 = x * x * x;
  return 0.5f * x * (1.0f + tanhf(0.7978845608028654f * (x + 0.044715f * x3)));
}

// ---------------------------------------------------------------------------
// TDM descriptors. g1 is built once per operand (tile shape + strides); per
// K-step only the 57-bit global address and the LDS buffer address change.
// Pad: pad_interval=16 DWORDs (64B), pad_amount=0 (1 DWORD) -> LDS row stride
// 68B = LDT ushorts, bank-conflict free.
// ---------------------------------------------------------------------------
__device__ __forceinline__ i32x8 tdm_make_g1(int K, int rows)
{
  unsigned td0 = (unsigned)K;            // tensor_dim0 (elements)
  unsigned td1 = 0x100000u;              // tensor_dim1 (large: no OOB clipping)
  i32x8 g1;
  g1[0] = (int)((1u << 16)               // data_size = 1 (2 bytes)
              | (1u << 20)               // pad_enable
              | (3u << 22));             // pad_interval = 16 DWORDs; pad_amount=0
  g1[1] = (int)((td0 & 0xFFFFu) << 16);  // atomic_barrier_addr=0 | td0[15:0]
  g1[2] = (int)(((td0 >> 16) & 0xFFFFu) | ((td1 & 0xFFFFu) << 16));
  g1[3] = (int)(((td1 >> 16) & 0xFFFFu) | (32u << 16));   // tile_dim0 = 32
  g1[4] = (int)(unsigned)rows;                            // tile_dim1; tile_dim2=0
  g1[5] = (int)(unsigned)K;                               // tensor_dim0_stride lo
  g1[6] = 0;
  g1[7] = 0;
  return g1;
}

__device__ __forceinline__ void tdm_issue(unsigned lds_addr, unsigned long long ga,
                                          i32x8 g1)
{
  u32x4 g0;
  g0[0] = 1u;                                            // count=1 user descriptor
  g0[1] = lds_addr;                                      // LDS byte address
  g0[2] = (unsigned)(ga & 0xFFFFFFFFu);                  // global_addr[31:0]
  g0[3] = (unsigned)((ga >> 32) & 0x1FFFFFFu)            // global_addr[56:32]
        | (2u << 30);                                    // type = 2 ("image")
  i32x4 z4 = {0, 0, 0, 0};
#if defined(__clang_major__) && (__clang_major__ >= 23)
  i32x8 z8 = {0, 0, 0, 0, 0, 0, 0, 0};
  __builtin_amdgcn_tensor_load_to_lds(g0, g1, z4, z4, z8, 0);
#else
  __builtin_amdgcn_tensor_load_to_lds(g0, g1, z4, z4, 0);
#endif
}

// ---------------------------------------------------------------------------
// WMMA GEMM with TDM-staged, double-buffered LDS tiles.
//   A:  bf16 [M,K] row-major (activations)
//   Bt: bf16 [N,K] row-major (weights pre-transposed/converted)
//   out[M,N] = act(A @ Bt^T + bias) + residual ; f32 or bf16 store.
// Block tile 64x128; 4 waves, each wave computes 32x64 via 2x4 WMMA tiles.
// Requires M%64==0, N%128==0, K%32==0 (true at every call site).
// ---------------------------------------------------------------------------
__global__ __launch_bounds__(GEMM_TB)
void wmma_gemm_tdm_kernel(const unsigned short* __restrict__ A,
                          const unsigned short* __restrict__ Bt,
                          const float* __restrict__ bias,
                          const float* __restrict__ residual,
                          void* __restrict__ out, int out_bf16, int act,
                          int M, int N, int K)
{
  __shared__ __align__(16) unsigned short As[2][BM * LDT];
  __shared__ __align__(16) unsigned short Bs[2][BN * LDT];

  const int tid  = threadIdx.x;
  const int wave = tid >> 5;
  const int lane = tid & 31;
  const int l    = lane & 15;
  const int hh   = lane >> 4;          // half-wave select per ISA VGPR layouts
  const int wrow = (wave >> 1) * 32;   // wave sub-tile inside 64x128 block
  const int wcol = (wave & 1) * 64;
  const int m0 = blockIdx.y * BM;
  const int n0 = blockIdx.x * BN;

  const unsigned ldsA[2] = { (unsigned)(size_t)(void*)&As[0][0],
                             (unsigned)(size_t)(void*)&As[1][0] };
  const unsigned ldsB[2] = { (unsigned)(size_t)(void*)&Bs[0][0],
                             (unsigned)(size_t)(void*)&Bs[1][0] };
  unsigned long long gaA = (unsigned long long)(size_t)(A  + (size_t)m0 * K);
  unsigned long long gaB = (unsigned long long)(size_t)(Bt + (size_t)n0 * K);
  const i32x8 g1A = tdm_make_g1(K, BM);
  const i32x8 g1B = tdm_make_g1(K, BN);

  v8f acc[2][4];
  const v8f vzero = {0.f, 0.f, 0.f, 0.f, 0.f, 0.f, 0.f, 0.f};
#pragma unroll
  for (int i = 0; i < 2; ++i)
#pragma unroll
    for (int j = 0; j < 4; ++j) acc[i][j] = vzero;

  // one K-step of fragment loads + 8 WMMAs on buffer `cur`
  auto do_tile = [&](int cur) {
    const unsigned short* Asl = &As[cur][0];
    const unsigned short* Bsl = &Bs[cur][0];
    v16bf afrag[2], bfrag[4];
#pragma unroll
    for (int i = 0; i < 2; ++i) {
      union { unsigned int u[8]; v16bf v; } ua;
      int mrow = wrow + i * 16 + l;
#pragma unroll
      for (int vv = 0; vv < 8; ++vv) {
        int kp = (vv < 4) ? (hh * 8 + 2 * vv) : (16 + hh * 8 + 2 * (vv - 4));
        ua.u[vv] = *reinterpret_cast<const unsigned int*>(&Asl[mrow * LDT + kp]);
      }
      afrag[i] = ua.v;
    }
#pragma unroll
    for (int j = 0; j < 4; ++j) {
      union { unsigned int u[8]; v16bf v; } ub;
      int ncol = wcol + j * 16 + l;
#pragma unroll
      for (int vv = 0; vv < 8; ++vv) {
        int kk = (vv < 4) ? (hh * 8 + 2 * vv) : (16 + hh * 8 + 2 * (vv - 4));
        ub.u[vv] = *reinterpret_cast<const unsigned int*>(&Bsl[ncol * LDT + kk]);
      }
      bfrag[j] = ub.v;
    }
#pragma unroll
    for (int i = 0; i < 2; ++i)
#pragma unroll
      for (int j = 0; j < 4; ++j)
        acc[i][j] = __builtin_amdgcn_wmma_f32_16x16x32_bf16(
            false, afrag[i], false, bfrag[j], (short)0, acc[i][j], false, false);
  };

  // prologue: DMA tile 0 into buffer 0 (wave 0 only; EXEC ignored by TDM)
  if (tid < 32) {
    tdm_issue(ldsA[0], gaA, g1A);
    tdm_issue(ldsB[0], gaB, g1B);
  }

  // steady state: always prefetch next slab, fence current (in-order TENSORcnt)
  int cur = 0;
  for (int kt = 0; kt < K - BK; kt += BK) {
    if (tid < 32) {
      tdm_issue(ldsA[cur ^ 1], gaA + 2u * BK, g1A);
      tdm_issue(ldsB[cur ^ 1], gaB + 2u * BK, g1B);
      __builtin_amdgcn_s_wait_tensorcnt((unsigned short)2);
    }
    __syncthreads();
    do_tile(cur);
    __syncthreads();   // all reads of `cur` done before it is refilled
    cur ^= 1;
    gaA += 2u * BK;
    gaB += 2u * BK;
  }
  // tail: last tile has no prefetch behind it
  if (tid < 32) __builtin_amdgcn_s_wait_tensorcnt((unsigned short)0);
  __syncthreads();
  do_tile(cur);

  // ---- epilogue: C/D f32 16x16 layout -> VGPR r holds M = r + 8*hh, N = l
#pragma unroll
  for (int i = 0; i < 2; ++i) {
#pragma unroll
    for (int j = 0; j < 4; ++j) {
#pragma unroll
      for (int r = 0; r < 8; ++r) {
        int m = m0 + wrow + i * 16 + 8 * hh + r;
        int n = n0 + wcol + j * 16 + l;
        float val = acc[i][j][r];
        if (bias)     val += bias[n];
        if (act == 1) val  = gelu_tanh(val);
        if (residual) val += residual[(size_t)m * N + n];
        if (out_bf16)
          ((unsigned short*)out)[(size_t)m * N + n] = f2bf(val);
        else
          ((float*)out)[(size_t)m * N + n] = val;
      }
    }
  }
}

// ---------------------------------------------------------------------------
// Weight prep: f32 [K,N] -> bf16 [N,K] (LDS-tiled transpose, coalesced both ways)
// ---------------------------------------------------------------------------
__global__ __launch_bounds__(256)
void cvt_transpose_bf16_kernel(const float* __restrict__ in, unsigned short* __restrict__ out,
                               int K, int N)
{
  __shared__ float tile[32][33];
  int k0 = blockIdx.y * 32, n0 = blockIdx.x * 32;
  int tx = threadIdx.x & 31, ty = threadIdx.x >> 5;     // 8 rows of 32
  for (int i = ty; i < 32; i += 8)
    tile[i][tx] = in[(size_t)(k0 + i) * N + n0 + tx];
  __syncthreads();
  for (int i = ty; i < 32; i += 8)
    out[(size_t)(n0 + i) * K + k0 + tx] = f2bf(tile[tx][i]);
}

// Plain f32 -> bf16 (embedding head: already [V,K]); 4 elements per thread.
__global__ __launch_bounds__(256)
void cvt_bf16_kernel(const float* __restrict__ in, unsigned short* __restrict__ out)
{
  size_t i = ((size_t)blockIdx.x * 256 + threadIdx.x) * 4;
  float4 f = *reinterpret_cast<const float4*>(in + i);
  out[i + 0] = f2bf(f.x); out[i + 1] = f2bf(f.y);
  out[i + 2] = f2bf(f.z); out[i + 3] = f2bf(f.w);
}

// ---------------------------------------------------------------------------
// Embedding gather: h[row,:] = embedding[x[row],:]
// ---------------------------------------------------------------------------
__global__ void embed_kernel(const int* __restrict__ x, const float* __restrict__ emb,
                             float* __restrict__ h, int dim)
{
  int row = blockIdx.y;
  int col = blockIdx.x * blockDim.x + threadIdx.x;
  h[(size_t)row * dim + col] = emb[(size_t)x[row] * dim + col];
}

// ---------------------------------------------------------------------------
// RMSNorm (per row) -> bf16 output for WMMA A operand
// ---------------------------------------------------------------------------
__global__ __launch_bounds__(256)
void rmsnorm_bf16_kernel(const float* __restrict__ x, const float* __restrict__ scale,
                         unsigned short* __restrict__ out, int dim)
{
  __shared__ float red[256];
  int row = blockIdx.x;
  const float* xr = x + (size_t)row * dim;
  float s = 0.f;
  for (int i = threadIdx.x; i < dim; i += 256) { float v = xr[i]; s += v * v; }
  red[threadIdx.x] = s;
  __syncthreads();
  for (int st = 128; st > 0; st >>= 1) {
    if (threadIdx.x < st) red[threadIdx.x] += red[threadIdx.x + st];
    __syncthreads();
  }
  float rstd = rsqrtf(red[0] / (float)dim + 1e-6f);
  for (int i = threadIdx.x; i < dim; i += 256)
    out[(size_t)row * dim + i] = f2bf(xr[i] * rstd * scale[i]);
}

// ---------------------------------------------------------------------------
// Gate-loop scan: per (batch, channel) complex recurrence over S, fused with
// the silu(g) gate; emits bf16 directly for the wo GEMM.
// ---------------------------------------------------------------------------
__global__ __launch_bounds__(256)
void gateloop_scan_kernel(const float* __restrict__ q, const float* __restrict__ k,
                          const float* __restrict__ v, const float* __restrict__ g,
                          const float* __restrict__ a, unsigned short* __restrict__ outbf,
                          int Slen, int dim)
{
  int gid = blockIdx.x * blockDim.x + threadIdx.x;   // 0 .. B*dim-1
  int b = gid / dim;
  int d = gid % dim;
  int twodim = 2 * dim;
  float yr = 0.f, yi = 0.f;
  for (int s = 0; s < Slen; ++s) {
    size_t row = (size_t)b * Slen + s;
    float ar = a[row * twodim + d];
    float ai = a[row * twodim + dim + d];
    float r  = sqrtf(ar * ar + ai * ai);
    float sg = 1.f / (1.f + expf(-r));
    float sc = sg / fmaxf(r, 1e-30f);     // magnitude*e^{i*phase} = (sig/|a|)*a
    float gr = ar * sc, gi = ai * sc;
    float kv = k[row * dim + d] * v[row * dim + d];
    float nyr = gr * yr - gi * yi + kv;
    float nyi = gr * yi + gi * yr;
    yr = nyr; yi = nyi;
    float o  = q[row * dim + d] * yr;
    float gg = g[row * dim + d];
    float gated = o * gg * (1.f / (1.f + expf(-gg)));
    outbf[row * dim + d] = f2bf(gated);
  }
}

// ---------------------------------------------------------------------------
// Host-side schedule
// ---------------------------------------------------------------------------
static inline void launch_gemm(hipStream_t s, const unsigned short* A, const unsigned short* Bt,
                               const float* bias, const float* res, void* out,
                               int out_bf16, int act, int M, int N, int K)
{
  dim3 grid(N / BN, M / BM);
  wmma_gemm_tdm_kernel<<<grid, GEMM_TB, 0, s>>>(A, Bt, bias, res, out, out_bf16, act,
                                                M, N, K);
}

static inline void launch_cvtT(hipStream_t s, const float* in, unsigned short* out,
                               int K, int N)
{
  cvt_transpose_bf16_kernel<<<dim3(N / 32, K / 32), 256, 0, s>>>(in, out, K, N);
}

extern "C" void kernel_launch(void* const* d_in, const int* in_sizes, int n_in,
                              void* d_out, int out_size, void* d_ws, size_t ws_size,
                              hipStream_t stream) {
  constexpr int Bsz = 2, Slen = 1024, DIM = 1024, DEPTH = 4, VOC = 32000, MULT = 4;
  constexpr int Mrows = Bsz * Slen;                  // 2048

  const int*   x         = (const int*)  d_in[0];
  const float* emb       = (const float*)d_in[1];
  const float* wq        = (const float*)d_in[2];
  const float* wk        = (const float*)d_in[3];
  const float* wv        = (const float*)d_in[4];
  const float* wa        = (const float*)d_in[5];
  const float* wg        = (const float*)d_in[6];
  const float* wo        = (const float*)d_in[7];
  const float* gl_scale  = (const float*)d_in[8];
  const float* ff_scale  = (const float*)d_in[9];
  const float* w1        = (const float*)d_in[10];
  const float* b1        = (const float*)d_in[11];
  const float* w2        = (const float*)d_in[12];
  const float* b2        = (const float*)d_in[13];
  const float* fin_scale = (const float*)d_in[14];
  float* logits = (float*)d_out;

  const size_t Mi = 1024u * 1024u;
  char* ws = (char*)d_ws;
  float* h   = (float*)(ws + 0 * Mi);                    // [2048,1024]  8 MiB
  float* qb  = (float*)(ws + 8 * Mi);                    // [2048,1024]  8 MiB
  float* kb  = (float*)(ws + 16 * Mi);                   // [2048,1024]  8 MiB
  float* vb  = (float*)(ws + 24 * Mi);                   // [2048,1024]  8 MiB
  float* gb  = (float*)(ws + 32 * Mi);                   // [2048,1024]  8 MiB
  float* ab  = (float*)(ws + 40 * Mi);                   // [2048,2048] 16 MiB
  unsigned short* bfA = (unsigned short*)(ws + 56 * Mi); // [2048,4096] bf16 16 MiB
  unsigned short* bfB = (unsigned short*)(ws + 72 * Mi); // [2048,4096] bf16 16 MiB
  unsigned short* wbf = (unsigned short*)(ws + 88 * Mi); // bf16 weight scratch 64 MiB

  // h = embedding[x]
  embed_kernel<<<dim3(DIM / 256, Mrows), 256, 0, stream>>>(x, emb, h, DIM);

  for (int lyr = 0; lyr < DEPTH; ++lyr) {
    const float* wq_l = wq + (size_t)lyr * DIM * DIM;
    const float* wk_l = wk + (size_t)lyr * DIM * DIM;
    const float* wv_l = wv + (size_t)lyr * DIM * DIM;
    const float* wa_l = wa + (size_t)lyr * DIM * 2 * DIM;
    const float* wg_l = wg + (size_t)lyr * DIM * DIM;
    const float* wo_l = wo + (size_t)lyr * DIM * DIM;
    const float* w1_l = w1 + (size_t)lyr * DIM * MULT * DIM;
    const float* b1_l = b1 + (size_t)lyr * MULT * DIM;
    const float* w2_l = w2 + (size_t)lyr * MULT * DIM * DIM;
    const float* b2_l = b2 + (size_t)lyr * DIM;

    // ---- gate-loop block
    rmsnorm_bf16_kernel<<<Mrows, 256, 0, stream>>>(h, gl_scale + (size_t)lyr * DIM, bfA, DIM);
    launch_cvtT(stream, wq_l, wbf, DIM, DIM);
    launch_gemm(stream, bfA, wbf, nullptr, nullptr, qb, 0, 0, Mrows, DIM, DIM);
    launch_cvtT(stream, wk_l, wbf, DIM, DIM);
    launch_gemm(stream, bfA, wbf, nullptr, nullptr, kb, 0, 0, Mrows, DIM, DIM);
    launch_cvtT(stream, wv_l, wbf, DIM, DIM);
    launch_gemm(stream, bfA, wbf, nullptr, nullptr, vb, 0, 0, Mrows, DIM, DIM);
    launch_cvtT(stream, wa_l, wbf, DIM, 2 * DIM);
    launch_gemm(stream, bfA, wbf, nullptr, nullptr, ab, 0, 0, Mrows, 2 * DIM, DIM);
    launch_cvtT(stream, wg_l, wbf, DIM, DIM);
    launch_gemm(stream, bfA, wbf, nullptr, nullptr, gb, 0, 0, Mrows, DIM, DIM);

    gateloop_scan_kernel<<<(Bsz * DIM) / 256, 256, 0, stream>>>(qb, kb, vb, gb, ab, bfA,
                                                                Slen, DIM);
    // h += y @ wo
    launch_cvtT(stream, wo_l, wbf, DIM, DIM);
    launch_gemm(stream, bfA, wbf, nullptr, h, h, 0, 0, Mrows, DIM, DIM);

    // ---- feed-forward block
    rmsnorm_bf16_kernel<<<Mrows, 256, 0, stream>>>(h, ff_scale + (size_t)lyr * DIM, bfA, DIM);
    launch_cvtT(stream, w1_l, wbf, DIM, MULT * DIM);
    launch_gemm(stream, bfA, wbf, b1_l, nullptr, bfB, 1, 1, Mrows, MULT * DIM, DIM);
    launch_cvtT(stream, w2_l, wbf, MULT * DIM, DIM);
    launch_gemm(stream, bfB, wbf, b2_l, h, h, 0, 0, Mrows, DIM, MULT * DIM);
  }

  // ---- final norm + tied-embedding head (logits = hn @ E^T, E already [V,K])
  rmsnorm_bf16_kernel<<<Mrows, 256, 0, stream>>>(h, fin_scale, bfA, DIM);
  cvt_bf16_kernel<<<((size_t)VOC * DIM) / 4 / 256, 256, 0, stream>>>(emb, wbf);
  launch_gemm(stream, bfA, wbf, nullptr, nullptr, logits, 0, 0, Mrows, VOC, DIM);
}
